// SAM_Spectral_41369124995768
// MI455X (gfx1250) — matrix operand order repaired and checked by reading, not compiled, not run on hardware
//
#include <hip/hip_runtime.h>
#include <hip/hip_bf16.h>

// ---------------------------------------------------------------------------
// SAM-Spectral transformer block chain for MI455X (gfx1250, wave32, WMMA).
// Channel-mixing 1x1 convs run as f16 WMMA GEMMs (v_wmma_f32_16x16x32_f16).
// Both A and B fragments are staged in LDS in fragment order so every
// fragment load is 2x ds_load_b128 (no scalar u16 gather / b16 packing).
// ---------------------------------------------------------------------------

#define CDIM   128
#define HWDIM  16384          // 128*128
#define BATCH  4
#define HEADS  8
#define NTILE  64             // output columns per workgroup
#define XPITCH 136            // padded K pitch (halves), 16B-aligned, no conflicts
#define BIGN   ((size_t)BATCH * CDIM * HWDIM)   // 8,388,608 floats per tensor

typedef _Float16 v16h __attribute__((ext_vector_type(16)));
typedef _Float16 h8   __attribute__((ext_vector_type(8)));
typedef float    v8f  __attribute__((ext_vector_type(8)));

__device__ __forceinline__ float gelu_f(float x) {
    return 0.5f * x * (1.0f + erff(x * 0.70710678118654752f));
}

// ---------------------------------------------------------------------------
// Channel LayerNorm: per (b, h, w) position, normalize over 128 channels.
// grid = BATCH*HWDIM/256, block = 256. Coalesced along hw.
// ---------------------------------------------------------------------------
__global__ void __launch_bounds__(256) ln_kernel(const float* __restrict__ X,
                                                 float* __restrict__ Y,
                                                 const float* __restrict__ w,
                                                 const float* __restrict__ b) {
    int idx = blockIdx.x * 256 + threadIdx.x;          // < BATCH*HWDIM
    int bb  = idx >> 14;
    int n   = idx & (HWDIM - 1);
    const float* xp = X + (size_t)bb * CDIM * HWDIM + n;
    float s = 0.f, ss = 0.f;
    for (int c = 0; c < CDIM; ++c) {
        float v = xp[(size_t)c * HWDIM];
        s += v; ss += v * v;
    }
    float mu   = s * (1.0f / CDIM);
    float var  = ss * (1.0f / CDIM) - mu * mu;
    float rstd = rsqrtf(var + 1e-5f);
    float* yp = Y + (size_t)bb * CDIM * HWDIM + n;
    for (int c = 0; c < CDIM; ++c) {
        float v = xp[(size_t)c * HWDIM];
        yp[(size_t)c * HWDIM] = (v - mu) * rstd * w[c] + b[c];
    }
}

// ---------------------------------------------------------------------------
// WMMA GEMM: OUT[b, m, n] (op=) sum_k W[m, k] * X[b, k, n]
//   M = K = 128, N tile = 64 per workgroup; 8 waves each own a 16-row M slice
//   and 4 accumulators (16x64). A fragment reused across the 4 N-subtiles.
//   flags: 1 = gelu on input, 2 = gelu on output, 4 = accumulate into OUT.
// grid = dim3(HWDIM/NTILE, BATCH), block = 256 (8 wave32 waves).
// ---------------------------------------------------------------------------
__global__ void __launch_bounds__(256) gemm128_kernel(
        const float* __restrict__ Wm, int ldw,
        const float* __restrict__ Xin, float* __restrict__ Out, int flags) {
    __shared__ __align__(16) _Float16 sW[CDIM * CDIM];      // 32 KB, M-major
    __shared__ __align__(16) _Float16 sX[NTILE][XPITCH];    // N-major, ~17 KB

    const int tid = threadIdx.x;
    const int b   = blockIdx.y;
    const int n0  = blockIdx.x * NTILE;

    // Stage weights (f32 -> f16), full 128x128, M-major (fragment order for A).
    for (int i = tid; i < CDIM * CDIM; i += 256) {
        int m = i >> 7, k = i & 127;
        sW[i] = (_Float16)Wm[(size_t)m * ldw + k];
    }
    // Stage activation tile [K=128][N=64], transposed to N-major (fragment
    // order for B): global reads coalesced along n, LDS row = all K of one n.
    const float* xb = Xin + (size_t)b * CDIM * HWDIM;
    for (int i = tid; i < CDIM * NTILE; i += 256) {
        int k = i >> 6, n = i & (NTILE - 1);
        float v = xb[(size_t)k * HWDIM + n0 + n];
        if (flags & 1) v = gelu_f(v);
        sX[n][k] = (_Float16)v;
    }
    __syncthreads();

    const int wave = tid >> 5;
    const int lane = tid & 31;
    const int hi   = lane >> 4;      // lane half (0: lanes 0-15, 1: 16-31)
    const int lo   = lane & 15;
    const int m0   = wave * 16;

    v8f acc[4] = {{}, {}, {}, {}};
    #pragma unroll
    for (int kk = 0; kk < CDIM; kk += 32) {
        // A fragment (16x32 f16), ISA 7.12.2: two 16B LDS loads per lane.
        const h8* wr = (const h8*)(&sW[(size_t)(m0 + lo) * CDIM + kk + hi * 8]);
        h8 a0 = wr[0];   // K = kk + hi*8 + (0..7)
        h8 a1 = wr[2];   // K = kk + 16 + hi*8 + (0..7)
        v16h a;
        #pragma unroll
        for (int i = 0; i < 8; ++i) { a[i] = a0[i]; a[8 + i] = a1[i]; }

        #pragma unroll
        for (int nt = 0; nt < 4; ++nt) {
            // B fragment (32x16 f16): lane = N, elements = same K order as A.
            const h8* xr = (const h8*)(&sX[nt * 16 + lo][kk + hi * 8]);
            h8 b0 = xr[0];   // K = kk + hi*8 + (0..7)
            h8 b1 = xr[2];   // K = kk + 16 + hi*8 + (0..7)
            v16h bf;
            #pragma unroll
            for (int i = 0; i < 8; ++i) { bf[i] = b0[i]; bf[8 + i] = b1[i]; }

            acc[nt] = __builtin_amdgcn_wmma_f32_16x16x32_f16(
                    /*neg_a=*/false, a, /*neg_b=*/false, bf,
                    /*c_mod=*/(short)0, acc[nt],
                    /*reuse_a=*/false, /*reuse_b=*/false);
        }
    }

    // D layout: VGPR r -> M = m0 + r + 8*hi, N = lo (per 16-col subtile).
    float* ob = Out + (size_t)b * CDIM * HWDIM;
    #pragma unroll
    for (int nt = 0; nt < 4; ++nt) {
        #pragma unroll
        for (int r = 0; r < 8; ++r) {
            int m = m0 + hi * 8 + r;
            size_t idx = (size_t)m * HWDIM + n0 + nt * 16 + lo;
            float v = acc[nt][r];
            if (flags & 2) v = gelu_f(v);
            if (flags & 4) v += ob[idx];
            ob[idx] = v;
        }
    }
}

// ---------------------------------------------------------------------------
// Depthwise 3x3 conv, SAME padding. grid = BIGN/256, block = 256.
// ---------------------------------------------------------------------------
__global__ void __launch_bounds__(256) dwconv_kernel(const float* __restrict__ IN,
                                                     float* __restrict__ OUT,
                                                     const float* __restrict__ w) {
    int idx = blockIdx.x * 256 + threadIdx.x;  // enumerates [b][c][hw]
    int n = idx & (HWDIM - 1);
    int c = (idx >> 14) & 127;
    int y = n >> 7, x = n & 127;
    const float* wp = w + c * 9;
    const float* ip = IN + ((size_t)idx - n);  // base of this (b,c) plane
    float acc = 0.f;
    #pragma unroll
    for (int ky = 0; ky < 3; ++ky) {
        int yy = y + ky - 1;
        if (yy < 0 || yy > 127) continue;
        #pragma unroll
        for (int kx = 0; kx < 3; ++kx) {
            int xx = x + kx - 1;
            if (xx < 0 || xx > 127) continue;
            acc += wp[ky * 3 + kx] * ip[yy * 128 + xx];
        }
    }
    OUT[idx] = acc;
}

// ---------------------------------------------------------------------------
// In-place L2 normalization of each (b, c) row over the 16384 spatial values.
// grid = BATCH*CDIM, block = 256.
// ---------------------------------------------------------------------------
__global__ void __launch_bounds__(256) l2norm_kernel(float* __restrict__ Q) {
    float* row = Q + (size_t)blockIdx.x * HWDIM;
    __shared__ float red[256];
    float s = 0.f;
    for (int n = threadIdx.x; n < HWDIM; n += 256) { float v = row[n]; s += v * v; }
    red[threadIdx.x] = s;
    __syncthreads();
    for (int off = 128; off > 0; off >>= 1) {
        if (threadIdx.x < off) red[threadIdx.x] += red[threadIdx.x + off];
        __syncthreads();
    }
    float inv = 1.0f / fmaxf(sqrtf(red[0]), 1e-12f);
    for (int n = threadIdx.x; n < HWDIM; n += 256) row[n] *= inv;
}

// ---------------------------------------------------------------------------
// Per-(b, head) attention: S = (Q K^T) * rescale, softmax rows, O = S V.
// Head dim = 16, spatial N = 16384. grid = BATCH*HEADS, block = 256.
// ---------------------------------------------------------------------------
#define ACHUNK 256
__global__ void __launch_bounds__(256) attn_kernel(const float* __restrict__ Q,
                                                   const float* __restrict__ Kt,
                                                   const float* __restrict__ V,
                                                   const float* __restrict__ resc,
                                                   float* __restrict__ O) {
    int bh = blockIdx.x;
    int b = bh >> 3, h = bh & 7;
    const size_t base = ((size_t)b * CDIM + h * 16) * HWDIM;
    const float* qb = Q + base;
    const float* kb = Kt + base;
    const float* vb = V + base;
    float* ob = O + base;

    __shared__ float sq[16][ACHUNK];
    __shared__ float sk[16][ACHUNK];
    __shared__ float sS[16][16];

    int t = threadIdx.x;
    int cc = t >> 4, dd = t & 15;   // each thread owns S[cc][dd]
    float acc = 0.f;
    for (int n0 = 0; n0 < HWDIM; n0 += ACHUNK) {
        for (int i = t; i < 16 * ACHUNK; i += 256) {
            int r = i / ACHUNK, col = i % ACHUNK;
            sq[r][col] = qb[(size_t)r * HWDIM + n0 + col];
            sk[r][col] = kb[(size_t)r * HWDIM + n0 + col];
        }
        __syncthreads();
        #pragma unroll 8
        for (int j = 0; j < ACHUNK; ++j) acc += sq[cc][j] * sk[dd][j];
        __syncthreads();
    }
    sS[cc][dd] = acc * resc[h];
    __syncthreads();
    if (t < 16) {
        float mx = -1e30f;
        for (int j = 0; j < 16; ++j) mx = fmaxf(mx, sS[t][j]);
        float e[16], sum = 0.f;
        for (int j = 0; j < 16; ++j) { e[j] = expf(sS[t][j] - mx); sum += e[j]; }
        float inv = 1.0f / sum;
        for (int j = 0; j < 16; ++j) sS[t][j] = e[j] * inv;
    }
    __syncthreads();
    for (int n = t; n < HWDIM; n += 256) {
        float vv[16];
        #pragma unroll
        for (int d = 0; d < 16; ++d) vv[d] = vb[(size_t)d * HWDIM + n];
        #pragma unroll
        for (int c2 = 0; c2 < 16; ++c2) {
            float o = 0.f;
            #pragma unroll
            for (int d = 0; d < 16; ++d) o += sS[c2][d] * vv[d];
            ob[(size_t)c2 * HWDIM + n] = o;
        }
    }
}

// ---------------------------------------------------------------------------
extern "C" void kernel_launch(void* const* d_in, const int* in_sizes, int n_in,
                              void* d_out, int out_size, void* d_ws, size_t ws_size,
                              hipStream_t stream) {
    (void)in_sizes; (void)n_in; (void)out_size; (void)ws_size;

    const float* x     = (const float*)d_in[0];
    const float* ln1w  = (const float*)d_in[1];
    const float* ln1b  = (const float*)d_in[2];
    const float* wq    = (const float*)d_in[3];
    const float* wk    = (const float*)d_in[4];
    const float* wv    = (const float*)d_in[5];
    const float* dwq   = (const float*)d_in[6];
    const float* dwk   = (const float*)d_in[7];
    const float* dwv   = (const float*)d_in[8];
    const float* resc  = (const float*)d_in[9];
    const float* wproj = (const float*)d_in[10];
    const float* ln2w  = (const float*)d_in[11];
    const float* ln2b  = (const float*)d_in[12];
    const float* f1a   = (const float*)d_in[13];
    const float* f1dw  = (const float*)d_in[14];
    const float* f2a   = (const float*)d_in[15];
    const float* f2dw  = (const float*)d_in[16];
    const float* fout  = (const float*)d_in[17];

    float* ws = (float*)d_ws;
    float* X  = ws;                 // residual stream
    float* Y  = ws + 1 * BIGN;      // LN output
    float* T  = ws + 2 * BIGN;      // scratch (conv1x1 out / attn out)
    float* Qb = ws + 3 * BIGN;
    float* Kb = ws + 4 * BIGN;
    float* Vb = ws + 5 * BIGN;

    const dim3 ggrid(HWDIM / NTILE, BATCH);  // 256 x 4 workgroups per GEMM
    const int  gblk = 256;
    const int  lnblocks = (BATCH * HWDIM) / 256;   // 256
    const int  dwblocks = (int)(BIGN / 256);       // 32768

    hipMemcpyAsync(X, x, BIGN * sizeof(float), hipMemcpyDeviceToDevice, stream);

    for (int i = 0; i < 2; ++i) {
        const size_t mo  = (size_t)i * CDIM * CDIM;
        const size_t lo  = (size_t)i * CDIM;
        const size_t dwo = (size_t)i * CDIM * 9;
        const size_t fo  = (size_t)i * CDIM * 2 * CDIM;

        // ---- Attention sub-block ----
        ln_kernel<<<lnblocks, 256, 0, stream>>>(X, Y, ln1w + lo, ln1b + lo);

        gemm128_kernel<<<ggrid, gblk, 0, stream>>>(wq + mo, CDIM, Y, T, 0);
        dwconv_kernel<<<dwblocks, 256, 0, stream>>>(T, Qb, dwq + dwo);
        gemm128_kernel<<<ggrid, gblk, 0, stream>>>(wk + mo, CDIM, Y, T, 0);
        dwconv_kernel<<<dwblocks, 256, 0, stream>>>(T, Kb, dwk + dwo);
        gemm128_kernel<<<ggrid, gblk, 0, stream>>>(wv + mo, CDIM, Y, T, 0);
        dwconv_kernel<<<dwblocks, 256, 0, stream>>>(T, Vb, dwv + dwo);

        l2norm_kernel<<<BATCH * CDIM, 256, 0, stream>>>(Qb);
        l2norm_kernel<<<BATCH * CDIM, 256, 0, stream>>>(Kb);

        attn_kernel<<<BATCH * HEADS, 256, 0, stream>>>(Qb, Kb, Vb,
                                                       resc + (size_t)i * HEADS, T);
        // projection + residual (accumulate into X)
        gemm128_kernel<<<ggrid, gblk, 0, stream>>>(wproj + mo, CDIM, T, X, 4);

        // ---- FFN sub-block ----
        ln_kernel<<<lnblocks, 256, 0, stream>>>(X, Y, ln2w + lo, ln2b + lo);

        gemm128_kernel<<<ggrid, gblk, 0, stream>>>(f1a + mo, CDIM, Y, T, 2); // gelu out
        dwconv_kernel<<<dwblocks, 256, 0, stream>>>(T, Qb, f1dw + dwo);      // o1
        gemm128_kernel<<<ggrid, gblk, 0, stream>>>(f2a + mo, CDIM, Y, T, 2); // gelu out
        dwconv_kernel<<<dwblocks, 256, 0, stream>>>(T, Kb, f2dw + dwo);      // o2

        // fout * gelu(cat(o1,o2)) + residual, split into two 128x128 GEMMs
        gemm128_kernel<<<ggrid, gblk, 0, stream>>>(fout + fo,        2 * CDIM, Qb, X, 1 | 4);
        gemm128_kernel<<<ggrid, gblk, 0, stream>>>(fout + fo + CDIM, 2 * CDIM, Kb, X, 1 | 4);
    }

    hipMemcpyAsync(d_out, X, BIGN * sizeof(float), hipMemcpyDeviceToDevice, stream);
}
